// HeteroICS_9405978378268
// MI455X (gfx1250) — matrix operand order repaired and checked by reading, not compiled, not run on hardware
//
#include <hip/hip_runtime.h>
#include <hip/hip_bf16.h>

typedef __attribute__((ext_vector_type(16))) _Float16 v16h;
typedef __attribute__((ext_vector_type(8)))  float    v8f;
typedef __attribute__((ext_vector_type(4)))  float    v4f;

#define NB_   256      // batch
#define NT_   64       // nodes per type
#define L_    64
#define D_    128
#define H_    8
#define KNN_  16
#define NN_   16384    // B*NT  (rows per node type)
#define ROWS_ 32768    // B*NUM_NODES

// ---------------------------------------------------------------------------
// WMMA GEMM, N fixed at 128:  C[M,128] = A[M,K] @ W[128,K]^T + bias
//   - Block = 128 threads = 4 waves. The block stages all of W into LDS as
//     f16 once; B fragments then come from LDS as aligned 32B reads.
//   - Each wave computes a 16x128 row strip: the A fragment is loaded ONCE
//     per k-chunk and reused across all 8 N-tiles (8 accumulators). This cuts
//     A traffic 8x vs a 16x16-per-wave scheme -- the key win since the big
//     GEMMs here are HBM-bound (A is 4-8MB, W is 32-64KB and L2-resident).
//   - A element (r,k) at A[(r/a_rb)*a_outer + (r%a_rb)*a_inner + k]
//     (a_rb=64/outer=8192/inner=64 addresses the strided x slices directly;
//      a_rb=1<<30 gives a plain row-major matrix).
// ---------------------------------------------------------------------------
__global__ __launch_bounds__(128)
void hgx_wmma_gemm(const float* __restrict__ A, int a_rb, long a_outer, int a_inner,
                   const float* __restrict__ W, const float* __restrict__ bias,
                   float* __restrict__ C, int ldc, int K)
{
    __shared__ _Float16 ldsW[128 * 128];           // up to 32KB (K<=128)

    // ---- cooperative stage: W[128][K] f32 -> LDS f16 (uniform trip count) --
    const int tid = threadIdx.x;
    const int tot = 128 * K;                       // 8192 or 16384, % 512 == 0
    for (int e = tid * 4; e < tot; e += 128 * 4) {
        const v4f w4 = *(const v4f*)(W + e);
        ldsW[e + 0] = (_Float16)w4.x;
        ldsW[e + 1] = (_Float16)w4.y;
        ldsW[e + 2] = (_Float16)w4.z;
        ldsW[e + 3] = (_Float16)w4.w;
    }
    __syncthreads();

    const int wave = tid >> 5;
    const int lane = tid & 31;
    const int half = lane >> 4;
    const int l15  = lane & 15;
    const long tileM = (long)blockIdx.x * 4 + wave;
    const long m     = tileM * 16 + l15;           // A row for this lane
    const float* arow = A + (m / a_rb) * a_outer + (m % a_rb) * (long)a_inner;

    v8f acc[8];
#pragma unroll
    for (int tn = 0; tn < 8; ++tn)
        acc[tn] = (v8f){0.f,0.f,0.f,0.f,0.f,0.f,0.f,0.f};

    for (int k0 = 0; k0 < K; k0 += 32) {
        __builtin_prefetch(arow + k0 + 32, 0, 0);
        // A 16x32 f16 fragment: lanes 0-15 hold K 0-15, lanes 16-31 K 16-31
        v16h a;
        const int ab = k0 + half * 16;
#pragma unroll
        for (int j = 0; j < 8; ++j) {
            a[j]     = (_Float16)arow[ab + j];
            a[j + 8] = (_Float16)arow[ab + 8 + j];
        }
        // reuse the A fragment across all 8 N-tiles
        const int bb = k0 + half * 16;             // 32B-aligned in LDS
#pragma unroll
        for (int tn = 0; tn < 8; ++tn) {
            const v16h b = *(const v16h*)(ldsW + (tn * 16 + l15) * K + bb);
            acc[tn] = __builtin_amdgcn_wmma_f32_16x16x32_f16(false, a, false, b,
                                                             (short)0, acc[tn],
                                                             false, false);
        }
    }
    // ---- epilogue: C/D fragment rows = half*8 + r, col = tn*16 + l15 ------
#pragma unroll
    for (int tn = 0; tn < 8; ++tn) {
        const int  n   = tn * 16 + l15;
        const float bs = bias ? bias[n] : 0.f;
#pragma unroll
        for (int r = 0; r < 8; ++r) {
            const long row = tileM * 16 + half * 8 + r;
            C[row * ldc + n] = acc[tn][r] + bs;
        }
    }
}

// ---------------------------------------------------------------------------
// LayerNorm of embedding table: v[node] = LN(emb[node])*g + b     (128 x 128)
// ---------------------------------------------------------------------------
__global__ void hgx_layernorm(const float* __restrict__ emb, const float* __restrict__ g,
                              const float* __restrict__ bb, float* __restrict__ v)
{
    __shared__ float red[128];
    const int node = blockIdx.x, c = threadIdx.x;
    const float x = emb[(long)node * D_ + c];
    red[c] = x; __syncthreads();
    for (int st = 64; st > 0; st >>= 1) { if (c < st) red[c] += red[c + st]; __syncthreads(); }
    const float mean = red[0] * (1.f / 128.f);
    __syncthreads();
    const float d = x - mean;
    red[c] = d * d; __syncthreads();
    for (int st = 64; st > 0; st >>= 1) { if (c < st) red[c] += red[c + st]; __syncthreads(); }
    const float var = red[0] * (1.f / 128.f);
    v[(long)node * D_ + c] = d * rsqrtf(var + 1e-5f) * g[c] + bb[c];
}

__global__ void hgx_vnorm(const float* __restrict__ v, float* __restrict__ vn)
{
    const int node = threadIdx.x;   // 128 threads
    float s = 0.f;
    for (int e = 0; e < D_; ++e) { const float t = v[(long)node * D_ + e]; s += t * t; }
    vn[node] = sqrtf(s);
}

// vbase[t][i][d] = v[idx_t(i)] . Wv_t[d] + bv_t[d]
__global__ void hgx_vb(const float* __restrict__ v,
                       const float* __restrict__ Wv_s, const float* __restrict__ bv_s,
                       const float* __restrict__ Wv_a, const float* __restrict__ bv_a,
                       float* __restrict__ vb)
{
    __shared__ float row[128];
    const int blk = blockIdx.x;           // t*64 + i
    const int t = blk >> 6, i = blk & 63;
    const int d = threadIdx.x;            // 128
    row[d] = v[(long)(t * 64 + i) * D_ + d];
    __syncthreads();
    const float* W  = t ? Wv_a : Wv_s;
    const float* bv = t ? bv_a : bv_s;
    float s = 0.f;
    for (int e = 0; e < D_; ++e) s += W[(long)d * D_ + e] * row[e];
    vb[(long)blk * D_ + d] = s + bv[d];
}

// ---------------------------------------------------------------------------
// Cosine-similarity top-K edges (batch-invariant): 4 edge types x 64 src rows
// ---------------------------------------------------------------------------
__global__ void hgx_topk(const float* __restrict__ v, const float* __restrict__ vn,
                         int* __restrict__ tk)
{
    __shared__ float sims[64][65];
    const int i = blockIdx.x;             // edge type
    const int s = threadIdx.x;            // src row 0..63
    const int soff = (i >> 1) * 64, doff = (i & 1) * 64;
    const float ns = vn[soff + s];
    for (int j = 0; j < 64; ++j) {
        float dot = 0.f;
        for (int e = 0; e < D_; ++e)
            dot += v[(long)(soff + s) * D_ + e] * v[(long)(doff + j) * D_ + e];
        sims[s][j] = dot / (ns * vn[doff + j]);
    }
    unsigned long long used = 0ull;
    for (int k = 0; k < KNN_; ++k) {
        float best = -3.4e38f; int bi = 0;
        for (int j = 0; j < 64; ++j)
            if (!((used >> j) & 1ull) && sims[s][j] > best) { best = sims[s][j]; bi = j; }
        used |= 1ull << bi;
        tk[((long)i * 64 + s) * KNN_ + k] = bi;
    }
}

// es[n][h] = sum_dh m[n][h*16+dh]*asrc[h][dh];  ed from xp_dst & adst
__global__ void hgx_esed(const float* __restrict__ m, const float* __restrict__ xpd,
                         const float* __restrict__ asrc, const float* __restrict__ adst,
                         float* __restrict__ es, float* __restrict__ ed)
{
    const int idx = blockIdx.x * blockDim.x + threadIdx.x;   // n*8+h
    if (idx >= NN_ * H_) return;
    const int n = idx >> 3, h = idx & 7;
    const long base = (long)n * D_ + h * 16;
    float se = 0.f, de = 0.f;
    for (int j = 0; j < 16; ++j) {
        se += m[base + j]   * asrc[h * 16 + j];
        de += xpd[base + j] * adst[h * 16 + j];
    }
    es[idx] = se; ed[idx] = de;
}

// ---------------------------------------------------------------------------
// GAT segment-softmax aggregation per destination node. Top-k indices within a
// row are distinct -> each src contributes <=1 edge to a given dst (<=64 edges).
// One block (128 threads = channels) per destination node.
// ---------------------------------------------------------------------------
__global__ void hgx_gat(const float* __restrict__ m, const float* __restrict__ es,
                        const float* __restrict__ ed, const int* __restrict__ tk,
                        float* __restrict__ out)
{
    __shared__ int   flag[64];
    __shared__ float mx[8], den[8];
    const int db = blockIdx.x;       // dst index b*64+dl
    const int b  = db >> 6;
    const int dl = db & 63;
    const int c  = threadIdx.x;      // channel 0..127
    const int h  = c >> 4;
    if (c < 64) {
        int f = 0;
        for (int k = 0; k < KNN_; ++k) f |= (tk[c * KNN_ + k] == dl);
        flag[c] = f;
    }
    __syncthreads();
    if (c < 8) {
        const float edv = ed[(long)db * H_ + c];
        float best = -3.4e38f;
        for (int s = 0; s < 64; ++s) if (flag[s]) {
            float lg = es[((long)b * 64 + s) * H_ + c] + edv;
            lg = lg >= 0.f ? lg : 0.2f * lg;
            best = fmaxf(best, lg);
        }
        mx[c] = best;
        float dsum = 0.f;
        for (int s = 0; s < 64; ++s) if (flag[s]) {
            float lg = es[((long)b * 64 + s) * H_ + c] + edv;
            lg = lg >= 0.f ? lg : 0.2f * lg;
            dsum += expf(lg - best);
        }
        den[c] = dsum;
    }
    __syncthreads();
    const float edv = ed[(long)db * H_ + h];
    const float mxh = mx[h], dnh = den[h] + 1e-16f;
    float o = 0.f;
    for (int s = 0; s < 64; ++s) if (flag[s]) {
        float lg = es[((long)b * 64 + s) * H_ + h] + edv;
        lg = lg >= 0.f ? lg : 0.2f * lg;
        o += (expf(lg - mxh) / dnh) * m[((long)b * 64 + s) * D_ + c];
    }
    out[(long)db * D_ + c] = o;
}

// sum over n,e of q[e]*tanh(proj[n][e])  -> 64 per-block partials (deterministic)
__global__ void hgx_semred(const float* __restrict__ proj, const float* __restrict__ q,
                           float* __restrict__ partial)
{
    __shared__ float red[256];
    const long total = (long)NN_ * D_;
    float s = 0.f;
    for (long idx = (long)blockIdx.x * blockDim.x + threadIdx.x; idx < total;
         idx += (long)gridDim.x * blockDim.x) {
        const int c = (int)(idx & 127);
        s += q[c] * tanhf(proj[idx]);
    }
    red[threadIdx.x] = s; __syncthreads();
    for (int st = 128; st > 0; st >>= 1) {
        if ((int)threadIdx.x < st) red[threadIdx.x] += red[threadIdx.x + st];
        __syncthreads();
    }
    if (threadIdx.x == 0) partial[blockIdx.x] = red[0];
}

// finalize semantic weights: beta[t][m] = softmax_m( w[t][m] )
__global__ void hgx_semfin(const float* __restrict__ wpart, float* __restrict__ beta)
{
    if (threadIdx.x == 0) {
        float w[4];
        for (int m = 0; m < 4; ++m) {
            float s = 0.f;
            for (int j = 0; j < 64; ++j) s += wpart[m * 64 + j];
            w[m] = s / (float)NN_;
        }
        for (int t = 0; t < 2; ++t) {
            const float w0 = w[t * 2], w1 = w[t * 2 + 1];
            const float mxv = fmaxf(w0, w1);
            const float e0 = expf(w0 - mxv), e1 = expf(w1 - mxv), d = e0 + e1;
            beta[t * 2]     = e0 / d;
            beta[t * 2 + 1] = e1 / d;
        }
    }
}

// p[b*128 + t*64 + i][c] = (beta0*zA + beta1*zB) * vb[t][i][c] + xp_t[n][c]
__global__ void hgx_combine(const float* __restrict__ zbuf, const float* __restrict__ beta,
                            const float* __restrict__ xps, const float* __restrict__ xpa,
                            const float* __restrict__ vb, float* __restrict__ p)
{
    const long idx = (long)blockIdx.x * blockDim.x + threadIdx.x;
    if (idx >= 2L * NN_ * D_) return;
    const int  t  = idx >= (long)NN_ * D_;
    const long li = idx - (long)t * NN_ * D_;        // n*128+c
    const int  n  = (int)(li >> 7);
    const int  c  = (int)(li & 127);
    const int  b  = n >> 6, i = n & 63;
    const float* zA = zbuf + (long)(t)     * NN_ * D_;   // edge types: sensor {0,2}, actuator {1,3}
    const float* zB = zbuf + (long)(2 + t) * NN_ * D_;
    const float z = beta[t * 2] * zA[li] + beta[t * 2 + 1] * zB[li];
    const float* xp = t ? xpa : xps;
    p[((long)b * 128 + t * 64 + i) * D_ + c] = z * vb[(long)t * 8192 + i * D_ + c] + xp[li];
}

// per-channel batch stats over 32768 rows
__global__ void hgx_bnstats(const float* __restrict__ p, float* __restrict__ mu,
                            float* __restrict__ var)
{
    __shared__ float rs[256], rq[256];
    const int c = blockIdx.x;
    float s = 0.f, q = 0.f;
    for (int n = threadIdx.x; n < ROWS_; n += 256) {
        const float v = p[(long)n * D_ + c];
        s += v; q += v * v;
    }
    rs[threadIdx.x] = s; rq[threadIdx.x] = q; __syncthreads();
    for (int st = 128; st > 0; st >>= 1) {
        if ((int)threadIdx.x < st) { rs[threadIdx.x] += rs[threadIdx.x + st];
                                     rq[threadIdx.x] += rq[threadIdx.x + st]; }
        __syncthreads();
    }
    if (threadIdx.x == 0) {
        const float m = rs[0] / (float)ROWS_;
        mu[c] = m;
        var[c] = rq[0] / (float)ROWS_ - m * m;
    }
}

// ---------------------------------------------------------------------------
// Fused tail: BN+LeakyReLU(0.01) on A-load -> WMMA x W1 -> ReLU -> dot W2 -> out
// One wave per 16 output rows; h[32768][256] never materialized.
// ---------------------------------------------------------------------------
__global__ __launch_bounds__(32)
void hgx_mlp(const float* __restrict__ P,
             const float* __restrict__ mu, const float* __restrict__ var,
             const float* __restrict__ bng, const float* __restrict__ bnb,
             const float* __restrict__ W1, const float* __restrict__ b1,
             const float* __restrict__ W2, const float* __restrict__ b2,
             float* __restrict__ out)
{
    __shared__ float part[16][17];
    const int lane = threadIdx.x, half = lane >> 4, l15 = lane & 15;
    const long m = (long)blockIdx.x * 16 + l15;
    const float* prow = P + m * D_;

    v16h afrag[4];
#pragma unroll
    for (int kc = 0; kc < 4; ++kc) {
        const int ab = kc * 32 + half * 16;
#pragma unroll
        for (int j = 0; j < 8; ++j) {
            const int kA = ab + j, kB = ab + 8 + j;
            float va = (prow[kA] - mu[kA]) * rsqrtf(var[kA] + 1e-5f) * bng[kA] + bnb[kA];
            float vb = (prow[kB] - mu[kB]) * rsqrtf(var[kB] + 1e-5f) * bng[kB] + bnb[kB];
            va = va > 0.f ? va : 0.01f * va;
            vb = vb > 0.f ? vb : 0.01f * vb;
            afrag[kc][j]     = (_Float16)va;
            afrag[kc][j + 8] = (_Float16)vb;
        }
    }
    float s[8] = {0.f,0.f,0.f,0.f,0.f,0.f,0.f,0.f};
    for (int tn = 0; tn < 16; ++tn) {           // 256 output features = 16 N-tiles
        const int n = tn * 16 + l15;
        const float* wrow = W1 + (long)n * D_;
        v8f acc = {0.f,0.f,0.f,0.f,0.f,0.f,0.f,0.f};
#pragma unroll
        for (int kc = 0; kc < 4; ++kc) {
            v16h bfr;
            const int bb = kc * 32 + half * 16;
#pragma unroll
            for (int j = 0; j < 16; ++j) bfr[j] = (_Float16)wrow[bb + j];
            acc = __builtin_amdgcn_wmma_f32_16x16x32_f16(false, afrag[kc], false, bfr,
                                                         (short)0, acc, false, false);
        }
        const float w2n = W2[n], b1n = b1[n];
#pragma unroll
        for (int r = 0; r < 8; ++r) s[r] += fmaxf(acc[r] + b1n, 0.f) * w2n;
    }
#pragma unroll
    for (int r = 0; r < 8; ++r) part[half * 8 + r][l15] = s[r];
    __syncthreads();
    if (lane < 16) {
        float t = 0.f;
#pragma unroll
        for (int c = 0; c < 16; ++c) t += part[lane][c];
        out[(long)blockIdx.x * 16 + lane] = t + b2[0];
    }
}

// ---------------------------------------------------------------------------
extern "C" void kernel_launch(void* const* d_in, const int* in_sizes, int n_in,
                              void* d_out, int out_size, void* d_ws, size_t ws_size,
                              hipStream_t stream)
{
    (void)in_sizes; (void)n_in; (void)out_size; (void)ws_size;
    const float* x    = (const float*)d_in[0];
    const float* emb  = (const float*)d_in[1];
    const float* ln_g = (const float*)d_in[2];
    const float* ln_b = (const float*)d_in[3];
    const float* bn_g = (const float*)d_in[4];
    const float* bn_b = (const float*)d_in[5];
    const float* W1   = (const float*)d_in[6];
    const float* b1   = (const float*)d_in[7];
    const float* W2   = (const float*)d_in[8];
    const float* b2   = (const float*)d_in[9];
    const float* Wx[2]   = {(const float*)d_in[10], (const float*)d_in[17]};
    const float* bx[2]   = {(const float*)d_in[11], (const float*)d_in[18]};
    const float* Wv[2]   = {(const float*)d_in[12], (const float*)d_in[19]};
    const float* bv[2]   = {(const float*)d_in[13], (const float*)d_in[20]};
    const float* Wsem[2] = {(const float*)d_in[14], (const float*)d_in[21]};
    const float* bsem[2] = {(const float*)d_in[15], (const float*)d_in[22]};
    const float* qv[2]   = {(const float*)d_in[16], (const float*)d_in[23]};
    const float* We[4]   = {(const float*)d_in[24], (const float*)d_in[27],
                            (const float*)d_in[30], (const float*)d_in[33]};
    const float* asrc[4] = {(const float*)d_in[25], (const float*)d_in[28],
                            (const float*)d_in[31], (const float*)d_in[34]};
    const float* adst[4] = {(const float*)d_in[26], (const float*)d_in[29],
                            (const float*)d_in[32], (const float*)d_in[35]};

    float* ws    = (float*)d_ws;
    float* v_    = ws;                       // 128*128
    float* vnorm = ws + 16384;               // 128
    float* vb    = ws + 16512;               // 2*64*128
    float* beta  = ws + 32896;               // 4
    float* wpart = ws + 32900;               // 4*64
    float* mu    = ws + 33156;               // 128
    float* var   = ws + 33284;               // 128
    int*   tk    = (int*)(ws + 33412);       // 4*64*16 ints
    float* xp    = ws + 40960;               // [2][NN][128]
    float* zbuf  = xp   + 2L * NN_ * D_;     // [4][NN][128]
    float* mbuf  = zbuf + 4L * NN_ * D_;     // [NN][128]  (reused per edge type & as proj)
    float* es    = mbuf + (long)NN_ * D_;    // [NN][8]
    float* ed    = es   + (long)NN_ * H_;    // [NN][8]
    float* p     = ed   + (long)NN_ * H_;    // [ROWS][128]
    float* out   = (float*)d_out;            // [ROWS]

    // node features
    hgx_layernorm<<<128, 128, 0, stream>>>(emb, ln_g, ln_b, v_);
    hgx_vnorm<<<1, 128, 0, stream>>>(v_, vnorm);
    hgx_vb<<<128, 128, 0, stream>>>(v_, Wv[0], bv[0], Wv[1], bv[1], vb);
    hgx_topk<<<4, 64, 0, stream>>>(v_, vnorm, tk);

    const dim3 gg(NN_ / 64);                 // 256 blocks x 4 waves = 16x128 strips
    // xp_t = x_slice @ Wx_t^T + bx_t   (strided A addressing over x[B,128,64])
    hgx_wmma_gemm<<<gg, 128, 0, stream>>>(x,         64, 8192, 64, Wx[0], bx[0],
                                          xp,                 D_, L_);
    hgx_wmma_gemm<<<gg, 128, 0, stream>>>(x + 64*64, 64, 8192, 64, Wx[1], bx[1],
                                          xp + (long)NN_*D_,  D_, L_);

    // per-edge-type: m = xp_src @ We^T ; es/ed ; GAT aggregation
    for (int i = 0; i < 4; ++i) {
        const int st = i >> 1, dt = i & 1;
        hgx_wmma_gemm<<<gg, 128, 0, stream>>>(xp + (long)st * NN_ * D_, 1 << 30, 0, D_,
                                              We[i], nullptr, mbuf, D_, D_);
        hgx_esed<<<512, 256, 0, stream>>>(mbuf, xp + (long)dt * NN_ * D_,
                                          asrc[i], adst[i], es, ed);
        hgx_gat<<<NN_, 128, 0, stream>>>(mbuf, es, ed, tk + i * 64 * KNN_,
                                         zbuf + (long)i * NN_ * D_);
    }

    // semantic attention: proj = z_part @ Wsem^T + bsem, reduce q.tanh(proj)
    for (int t = 0; t < 2; ++t)
        for (int mm = 0; mm < 2; ++mm) {
            const float* zp = zbuf + (long)(2 * mm + t) * NN_ * D_;
            hgx_wmma_gemm<<<gg, 128, 0, stream>>>(zp, 1 << 30, 0, D_,
                                                  Wsem[t], bsem[t], mbuf, D_, D_);
            hgx_semred<<<64, 256, 0, stream>>>(mbuf, qv[t], wpart + (t * 2 + mm) * 64);
        }
    hgx_semfin<<<1, 1, 0, stream>>>(wpart, beta);

    // combine z*vp + xp into p, then BN stats, then fused BN+MLP tail
    hgx_combine<<<16384, 256, 0, stream>>>(zbuf, beta, xp, xp + (long)NN_ * D_, vb, p);
    hgx_bnstats<<<128, 256, 0, stream>>>(p, mu, var);
    hgx_mlp<<<ROWS_ / 16, 32, 0, stream>>>(p, mu, var, bn_g, bn_b, W1, b1, W2, b2, out);
}
